// LiteEval_8366596293149
// MI455X (gfx1250) — matrix operand order repaired and checked by reading, not compiled, not run on hardware
//
#include <hip/hip_runtime.h>
#include <math.h>

typedef __bf16 bf16;
typedef __attribute__((ext_vector_type(16))) __bf16 v16bf;
typedef __attribute__((ext_vector_type(8)))  float  v8f;

#define BATCH 256
#define TSTEPS 64
#define LH 1024
#define SH 512
#define NCLS 239

__device__ __forceinline__ float sigm(float x) { return 1.0f / (1.0f + expf(-x)); }

// ---------------------------------------------------------------------------
// WMMA fragment loaders (CDNA5 16-bit layouts, wave32)
// A (16x32, MxK): lane L -> row (L&15); K elems {(L>>4)*8 + [0..8), 16+(L>>4)*8 + [0..8)}
// B (32x16, KxN): lane L -> col (L&15); K elems {(L>>4)*8 + [0..8), 16+(L>>4)*8 + [0..8)}
// ---------------------------------------------------------------------------
__device__ __forceinline__ v16bf load_frag_a_bf16(const bf16* __restrict__ A, int lda,
                                                  int mBase, int k0, int lane) {
  int m  = mBase + (lane & 15);
  int ka = k0 + ((lane >> 4) << 3);
  const bf16* p = A + (size_t)m * lda + ka;
  union { uint4 q[2]; v16bf v; } u;
  u.q[0] = *(const uint4*)(p);
  u.q[1] = *(const uint4*)(p + 16);
  return u.v;
}

__device__ __forceinline__ v16bf load_frag_a_f32(const float* __restrict__ A, int lda,
                                                 int mBase, int k0, int lane) {
  int m  = mBase + (lane & 15);
  int ka = k0 + ((lane >> 4) << 3);
  const float* p = A + (size_t)m * lda + ka;
  const float4* q = (const float4*)p;
  float4 f0 = q[0], f1 = q[1];   // k .. k+7
  float4 f2 = q[4], f3 = q[5];   // k+16 .. k+23
  v16bf v;
  v[0]=(bf16)f0.x; v[1]=(bf16)f0.y; v[2]=(bf16)f0.z; v[3]=(bf16)f0.w;
  v[4]=(bf16)f1.x; v[5]=(bf16)f1.y; v[6]=(bf16)f1.z; v[7]=(bf16)f1.w;
  v[8]=(bf16)f2.x; v[9]=(bf16)f2.y; v[10]=(bf16)f2.z; v[11]=(bf16)f2.w;
  v[12]=(bf16)f3.x; v[13]=(bf16)f3.y; v[14]=(bf16)f3.z; v[15]=(bf16)f3.w;
  return v;
}

__device__ __forceinline__ v16bf load_frag_b(const bf16* __restrict__ W, int ldw,
                                             int nBase, int k0, int lane) {
  int n  = nBase + (lane & 15);
  int kb = k0 + ((lane >> 4) << 3);
  const bf16* p = W + (size_t)n * ldw + kb;
  union { uint4 q[2]; v16bf v; } u;
  u.q[0] = *(const uint4*)(p);
  u.q[1] = *(const uint4*)(p + 16);
  return u.v;
}

// ---------------------------------------------------------------------------
// Generic WMMA GEMM: C[M,N] = A[M,K] @ W[N,K]^T, epilogue per mode.
// blockDim = 128 (4 waves); each wave computes a (16*MT)x32 tile:
//   MT=1: 2 WMMAs / k-step (1 A-frag + 2 B-frags)  -> max parallelism (scan GEMMs)
//   MT=2: 4 WMMAs / k-step (2 A-frags + 2 B-frags) -> +50% arith intensity (projections)
// grid = (N/128, M/(16*MT)).
// EPI 0: out_bf16 = relu(acc*scale[n] + shift[n])   (BN folded, bias folded into shift)
// EPI 1: out_f32  = acc + shift[n]                  (combined LSTM biases)
// ---------------------------------------------------------------------------
template <bool AF32, int EPI, int MT>
__global__ __launch_bounds__(128)
void gemm_wmma(const void* __restrict__ Aptr, const bf16* __restrict__ W,
               const float* __restrict__ scale, const float* __restrict__ shift,
               bf16* __restrict__ outB, float* __restrict__ outF,
               int M, int N, int K) {
  int lane  = threadIdx.x & 31;
  int wave  = threadIdx.x >> 5;
  int mBase = blockIdx.y * (16 * MT);
  int n0    = (blockIdx.x * 4 + wave) * 32;
  if (n0 >= N || mBase >= M) return;

  v8f acc00 = {0.f,0.f,0.f,0.f,0.f,0.f,0.f,0.f};
  v8f acc01 = acc00, acc10 = acc00, acc11 = acc00;

  for (int k0 = 0; k0 < K; k0 += 32) {
    v16bf a0, a1;
    if constexpr (AF32) {
      a0 = load_frag_a_f32((const float*)Aptr, K, mBase, k0, lane);
      if constexpr (MT == 2)
        a1 = load_frag_a_f32((const float*)Aptr, K, mBase + 16, k0, lane);
    } else {
      a0 = load_frag_a_bf16((const bf16*)Aptr, K, mBase, k0, lane);
      if constexpr (MT == 2)
        a1 = load_frag_a_bf16((const bf16*)Aptr, K, mBase + 16, k0, lane);
    }
    v16bf b0 = load_frag_b(W, K, n0,      k0, lane);
    v16bf b1 = load_frag_b(W, K, n0 + 16, k0, lane);
    acc00 = __builtin_amdgcn_wmma_f32_16x16x32_bf16(false, a0, false, b0, (short)0, acc00, false, false);
    acc01 = __builtin_amdgcn_wmma_f32_16x16x32_bf16(false, a0, false, b1, (short)0, acc01, false, false);
    if constexpr (MT == 2) {
      acc10 = __builtin_amdgcn_wmma_f32_16x16x32_bf16(false, a1, false, b0, (short)0, acc10, false, false);
      acc11 = __builtin_amdgcn_wmma_f32_16x16x32_bf16(false, a1, false, b1, (short)0, acc11, false, false);
    }
  }

  // C layout: VGPR j, lanes 0-15 -> row j, lanes 16-31 -> row j+8; col = lane&15
  int row0 = mBase + ((lane >> 4) << 3);
  int c0 = n0 + (lane & 15);
  int c1 = c0 + 16;
  float s0 = 0.f, s1 = 0.f;
  if constexpr (EPI == 0) { s0 = scale[c0]; s1 = scale[c1]; }
  float h0 = shift[c0], h1 = shift[c1];
#pragma unroll
  for (int j = 0; j < 8; j++) {
    size_t row = (size_t)(row0 + j);
    if constexpr (EPI == 0) {
      outB[row * N + c0] = (bf16)fmaxf(acc00[j] * s0 + h0, 0.0f);
      outB[row * N + c1] = (bf16)fmaxf(acc01[j] * s1 + h1, 0.0f);
      if constexpr (MT == 2) {
        outB[(row + 16) * N + c0] = (bf16)fmaxf(acc10[j] * s0 + h0, 0.0f);
        outB[(row + 16) * N + c1] = (bf16)fmaxf(acc11[j] * s1 + h1, 0.0f);
      }
    } else {
      outF[row * N + c0] = acc00[j] + h0;
      outF[row * N + c1] = acc01[j] + h1;
      if constexpr (MT == 2) {
        outF[(row + 16) * N + c0] = acc10[j] + h0;
        outF[(row + 16) * N + c1] = acc11[j] + h1;
      }
    }
  }
}

// ---------------------------------------------------------------------------
// Weight / vector prep kernels
// ---------------------------------------------------------------------------
__global__ void cvt_rows_bf16(bf16* __restrict__ dst, int dstride, int doff,
                              const float* __restrict__ src, int sstride,
                              int rows, int cols) {
  int idx = blockIdx.x * blockDim.x + threadIdx.x;
  if (idx >= rows * cols) return;
  int r = idx / cols, c = idx - r * cols;
  dst[(size_t)r * dstride + doff + c] = (bf16)src[(size_t)r * sstride + c];
}

__global__ void bn_prep(float* __restrict__ scale, float* __restrict__ shift,
                        const float* __restrict__ g, const float* __restrict__ be,
                        const float* __restrict__ bias, int n) {
  int i = blockIdx.x * blockDim.x + threadIdx.x;
  if (i >= n) return;
  float s = g[i] * (1.0f / sqrtf(1.0f + 1e-5f));
  scale[i] = s;
  shift[i] = bias[i] * s + be[i];
}

__global__ void vec_add(float* __restrict__ o, const float* __restrict__ a,
                        const float* __restrict__ b, int n) {
  int i = blockIdx.x * blockDim.x + threadIdx.x;
  if (i < n) o[i] = a[i] + b[i];
}

__global__ void zero_f32(float* __restrict__ p, int n) {
  int i = blockIdx.x * blockDim.x + threadIdx.x;
  if (i < n) p[i] = 0.0f;
}

// ---------------------------------------------------------------------------
// Per-step kernels
// ---------------------------------------------------------------------------
// Pack A_s[256,1024] = [s_t | bf16(h_s)],  A_l[256,3584] = [f_t | s_t | bf16(h_l)]
__global__ void pack_step(const bf16* __restrict__ feats, const bf16* __restrict__ sfeats,
                          const float* __restrict__ h_s, const float* __restrict__ h_l,
                          bf16* __restrict__ A_s, bf16* __restrict__ A_l, int t) {
  const int TOT_S = BATCH * 1024;
  const int TOT_L = BATCH * 3584;
  int idx = blockIdx.x * blockDim.x + threadIdx.x;
  if (idx < TOT_S) {
    int b = idx >> 10, j = idx & 1023;
    bf16 v;
    if (j < SH) v = sfeats[((size_t)b * TSTEPS + t) * SH + j];
    else        v = (bf16)h_s[b * SH + (j - SH)];
    A_s[idx] = v;
  } else if (idx < TOT_S + TOT_L) {
    int k = idx - TOT_S;
    int b = k / 3584, j = k - b * 3584;
    bf16 v;
    if (j < 2048)      v = feats[((size_t)b * TSTEPS + t) * 2048 + j];
    else if (j < 2560) v = sfeats[((size_t)b * TSTEPS + t) * SH + (j - 2048)];
    else               v = (bf16)h_l[b * LH + (j - 2560)];
    A_l[(size_t)b * 3584 + j] = v;
  }
}

// Gate: r = one_hot(argmax([s_t, h_l, c_l] @ W_g^T + b_g))  (STE forward value)
__global__ void gate_kernel(const bf16* __restrict__ sfeats, const float* __restrict__ h_l,
                            const float* __restrict__ c_l, const float* __restrict__ W_g,
                            const float* __restrict__ b_g, float* __restrict__ r_out,
                            float* __restrict__ r_buf, int t) {
  __shared__ float s0[256], s1[256];
  int b = blockIdx.x, tid = threadIdx.x;
  float a0 = 0.f, a1 = 0.f;
  for (int k = tid; k < 2560; k += 256) {
    float v;
    if (k < SH)          v = (float)sfeats[((size_t)b * TSTEPS + t) * SH + k];
    else if (k < SH+LH)  v = h_l[b * LH + (k - SH)];
    else                 v = c_l[b * LH + (k - SH - LH)];
    a0 += v * W_g[k];
    a1 += v * W_g[2560 + k];
  }
  s0[tid] = a0; s1[tid] = a1;
  __syncthreads();
  for (int s = 128; s > 0; s >>= 1) {
    if (tid < s) { s0[tid] += s0[tid + s]; s1[tid] += s1[tid + s]; }
    __syncthreads();
  }
  if (tid == 0) {
    float l0 = s0[0] + b_g[0], l1 = s1[0] + b_g[1];
    float r0 = (l0 >= l1) ? 1.0f : 0.0f;   // argmax ties -> index 0 (jnp.argmax)
    r_out[((size_t)t * BATCH + b) * 2 + 0] = r0;
    r_out[((size_t)t * BATCH + b) * 2 + 1] = 1.0f - r0;
    r_buf[b * 2 + 0] = r0;
    r_buf[b * 2 + 1] = 1.0f - r0;
  }
}

__global__ void small_cell(const float* __restrict__ gates,
                           float* __restrict__ h, float* __restrict__ c) {
  int idx = blockIdx.x * blockDim.x + threadIdx.x;
  if (idx >= BATCH * SH) return;
  int b = idx >> 9, j = idx & (SH - 1);
  const float* g = gates + (size_t)b * (4 * SH);
  float gi = g[j], gf = g[SH + j], gg = g[2 * SH + j], go = g[3 * SH + j];
  float c2 = sigm(gf) * c[idx] + sigm(gi) * tanhf(gg);
  float h2 = sigm(go) * tanhf(c2);
  c[idx] = c2;
  h[idx] = h2;
}

__global__ void large_cell(const float* __restrict__ gates, const float* __restrict__ h_s,
                           const float* __restrict__ c_s, const float* __restrict__ r_buf,
                           float* __restrict__ h, float* __restrict__ c) {
  int idx = blockIdx.x * blockDim.x + threadIdx.x;
  if (idx >= BATCH * LH) return;
  int b = idx >> 10, j = idx & (LH - 1);
  const float* g = gates + (size_t)b * (4 * LH);
  float gi = g[j], gf = g[LH + j], gg = g[2 * LH + j], go = g[3 * LH + j];
  float hold = h[idx], cold = c[idx];
  float c_ln = sigm(gf) * cold + sigm(gi) * tanhf(gg);
  float h_ln = sigm(go) * tanhf(c_ln);
  float ph = (j < SH) ? h_s[b * SH + j] : hold;
  float pc = (j < SH) ? c_s[b * SH + j] : cold;
  float r0 = r_buf[b * 2], r1 = r_buf[b * 2 + 1];
  h[idx] = r0 * h_ln + r1 * ph;
  c[idx] = r0 * c_ln + r1 * pc;
}

__global__ void classifier(const float* __restrict__ h, const float* __restrict__ Wc,
                           const float* __restrict__ bc, float* __restrict__ out) {
  int idx = blockIdx.x * blockDim.x + threadIdx.x;
  if (idx >= BATCH * NCLS) return;
  int b = idx / NCLS, n = idx - b * NCLS;
  const float* hr = h + (size_t)b * LH;
  const float* wr = Wc + (size_t)n * LH;
  float acc = bc[n];
  for (int k = 0; k < LH; k++) acc += hr[k] * wr[k];
  out[idx] = acc;
}

// ---------------------------------------------------------------------------
extern "C" void kernel_launch(void* const* d_in, const int* in_sizes, int n_in,
                              void* d_out, int out_size, void* d_ws, size_t ws_size,
                              hipStream_t stream) {
  // inputs (setup_inputs order)
  const float* x      = (const float*)d_in[0];   // [256,64,2048]
  const float* xs     = (const float*)d_in[1];   // [256,64,1280]
  const float* W_p    = (const float*)d_in[2];
  const float* b_p    = (const float*)d_in[3];
  const float* g_p    = (const float*)d_in[4];
  const float* be_p   = (const float*)d_in[5];
  const float* W_s    = (const float*)d_in[6];
  const float* b_s    = (const float*)d_in[7];
  const float* g_s    = (const float*)d_in[8];
  const float* be_s   = (const float*)d_in[9];
  const float* Wih_l  = (const float*)d_in[10];  // [4096,2560]
  const float* Whh_l  = (const float*)d_in[11];  // [4096,1024]
  const float* bih_l  = (const float*)d_in[12];
  const float* bhh_l  = (const float*)d_in[13];
  const float* Wih_s  = (const float*)d_in[14];  // [2048,512]
  const float* Whh_s  = (const float*)d_in[15];  // [2048,512]
  const float* bih_s  = (const float*)d_in[16];
  const float* bhh_s  = (const float*)d_in[17];
  const float* W_g    = (const float*)d_in[18];  // [2,2560]
  const float* b_g    = (const float*)d_in[19];
  const float* W_c    = (const float*)d_in[20];  // [239,1024]
  const float* b_c    = (const float*)d_in[21];

  float* logits = (float*)d_out;                      // [256,239]
  float* rstack = (float*)d_out + BATCH * NCLS;       // [64,256,2]

  // workspace bump allocator
  char* wsp = (char*)d_ws;
  auto alloc = [&](size_t bytes) -> void* {
    void* p = wsp;
    wsp += (bytes + 255) & ~(size_t)255;
    return p;
  };
  const int M = BATCH * TSTEPS;  // 16384
  bf16* feats_bf  = (bf16*)alloc((size_t)M * 2048 * 2);
  bf16* sfeats_bf = (bf16*)alloc((size_t)M * SH * 2);
  bf16* Wp_bf     = (bf16*)alloc((size_t)2048 * 2048 * 2);
  bf16* Ws_bf     = (bf16*)alloc((size_t)512 * 1280 * 2);
  bf16* Wcs_bf    = (bf16*)alloc((size_t)2048 * 1024 * 2);     // [Wih_s | Whh_s]
  bf16* Wcl_bf    = (bf16*)alloc((size_t)4096 * 3584 * 2);     // [Wih_l | Whh_l]
  bf16* A_s       = (bf16*)alloc((size_t)BATCH * 1024 * 2);
  bf16* A_l       = (bf16*)alloc((size_t)BATCH * 3584 * 2);
  float* gates_s  = (float*)alloc((size_t)BATCH * 4 * SH * 4);
  float* gates_l  = (float*)alloc((size_t)BATCH * 4 * LH * 4);
  float* states   = (float*)alloc((size_t)(2 * BATCH * SH + 2 * BATCH * LH) * 4);
  float* h_s = states;
  float* c_s = h_s + BATCH * SH;
  float* h_l = c_s + BATCH * SH;
  float* c_l = h_l + BATCH * LH;
  float* scale_p = (float*)alloc(2048 * 4);
  float* shift_p = (float*)alloc(2048 * 4);
  float* scale_s = (float*)alloc(SH * 4);
  float* shift_s = (float*)alloc(SH * 4);
  float* bcs     = (float*)alloc(4 * SH * 4);
  float* bcl     = (float*)alloc(4 * LH * 4);
  float* r_buf   = (float*)alloc(BATCH * 2 * 4);

  // ---- prep: fold BN, combine biases, convert weights to bf16 ----
  bn_prep<<<(2048 + 255) / 256, 256, 0, stream>>>(scale_p, shift_p, g_p, be_p, b_p, 2048);
  bn_prep<<<(SH + 255) / 256, 256, 0, stream>>>(scale_s, shift_s, g_s, be_s, b_s, SH);
  vec_add<<<(4 * SH + 255) / 256, 256, 0, stream>>>(bcs, bih_s, bhh_s, 4 * SH);
  vec_add<<<(4 * LH + 255) / 256, 256, 0, stream>>>(bcl, bih_l, bhh_l, 4 * LH);

  auto cvt = [&](bf16* dst, int dstride, int doff, const float* src, int sstride,
                 int rows, int cols) {
    int n = rows * cols;
    cvt_rows_bf16<<<(n + 255) / 256, 256, 0, stream>>>(dst, dstride, doff, src, sstride,
                                                       rows, cols);
  };
  cvt(Wp_bf, 2048, 0, W_p, 2048, 2048, 2048);
  cvt(Ws_bf, 1280, 0, W_s, 1280, 512, 1280);
  cvt(Wcs_bf, 1024, 0,   Wih_s, 512,  2048, 512);
  cvt(Wcs_bf, 1024, 512, Whh_s, 512,  2048, 512);
  cvt(Wcl_bf, 3584, 0,    Wih_l, 2560, 4096, 2560);
  cvt(Wcl_bf, 3584, 2560, Whh_l, 1024, 4096, 1024);

  int nstate = 2 * BATCH * SH + 2 * BATCH * LH;
  zero_f32<<<(nstate + 255) / 256, 256, 0, stream>>>(states, nstate);

  // ---- time-parallel projections (WMMA MT=2, BN+ReLU epilogue, bf16 out) ----
  gemm_wmma<true, 0, 2><<<dim3(2048 / 128, M / 32), 128, 0, stream>>>(
      x, Wp_bf, scale_p, shift_p, feats_bf, nullptr, M, 2048, 2048);
  gemm_wmma<true, 0, 2><<<dim3(SH / 128, M / 32), 128, 0, stream>>>(
      xs, Ws_bf, scale_s, shift_s, sfeats_bf, nullptr, M, SH, 1280);

  // ---- sequential scan (WMMA MT=1 for max wave parallelism per step) ----
  const int PACK_N = BATCH * 1024 + BATCH * 3584;
  for (int t = 0; t < TSTEPS; t++) {
    pack_step<<<(PACK_N + 255) / 256, 256, 0, stream>>>(feats_bf, sfeats_bf, h_s, h_l,
                                                        A_s, A_l, t);
    gate_kernel<<<BATCH, 256, 0, stream>>>(sfeats_bf, h_l, c_l, W_g, b_g, rstack, r_buf, t);
    gemm_wmma<false, 1, 1><<<dim3((4 * SH) / 128, BATCH / 16), 128, 0, stream>>>(
        A_s, Wcs_bf, nullptr, bcs, nullptr, gates_s, BATCH, 4 * SH, 1024);
    gemm_wmma<false, 1, 1><<<dim3((4 * LH) / 128, BATCH / 16), 128, 0, stream>>>(
        A_l, Wcl_bf, nullptr, bcl, nullptr, gates_l, BATCH, 4 * LH, 3584);
    small_cell<<<(BATCH * SH + 255) / 256, 256, 0, stream>>>(gates_s, h_s, c_s);
    large_cell<<<(BATCH * LH + 255) / 256, 256, 0, stream>>>(gates_l, h_s, c_s, r_buf,
                                                             h_l, c_l);
  }

  // ---- classifier ----
  classifier<<<(BATCH * NCLS + 255) / 256, 256, 0, stream>>>(h_l, W_c, b_c, logits);
}
